// MyModel_90640989815557
// MI455X (gfx1250) — matrix operand order repaired and checked by reading, not compiled, not run on hardware
//
#include <hip/hip_runtime.h>
#include <math.h>

// ---------------------------------------------------------------------------
// MI455X (gfx1250) implementation.
// Roofline: dominated by the single ~100MB read of qd_last_hidden (~4.3us at
// 23.3 TB/s) -> pooling is split across 512 blocks (two-pass, deterministic).
// GEMMs (~1 GFLOP total) run through v_wmma_f32_16x16x32_f16 with fp32
// accumulation; each wave computes a 16x32 strip so one A fragment feeds two
// WMMAs. Graph attention uses the compile-time 28-edge adjacency, fully
// unrolled.
// ---------------------------------------------------------------------------

#define BQ   64   // batch
#define SQ   512  // sequence
#define HQ   768  // hidden
#define NPG_ 8    // nodes per subgraph
#define NNODES 512   // BQ * NPG_
#define HEADS1_ 4
#define C1_    128
#define H1DIM  512   // HEADS1_*C1_
#define NCHUNK 8     // sequence chunks for pooling
#define CHLEN  (SQ / NCHUNK)

typedef __attribute__((ext_vector_type(16))) _Float16 v16h;
typedef __attribute__((ext_vector_type(8)))  _Float16 v8h;
typedef __attribute__((ext_vector_type(8)))  float    v8f;

// ------------------------------ utilities ----------------------------------

__device__ inline float wave_reduce_sum(float v) {
  #pragma unroll
  for (int off = 16; off > 0; off >>= 1) v += __shfl_xor(v, off, 32);
  return v;
}

// ------------------------ 1. masked mean pooling ---------------------------
// Pass 1: grid (B, NCHUNK); each block accumulates its sequence chunk.
__global__ void pool_partial(const float* __restrict__ qd,
                             const int* __restrict__ sep1,
                             const int* __restrict__ sep2,
                             float* __restrict__ partQ, float* __restrict__ partD) {
  int b = blockIdx.x, c = blockIdx.y;
  int h = threadIdx.x;                       // 768 threads, coalesced along H
  int s1 = sep1[b], s2 = sep2[b];
  int sbeg = c * CHLEN;
  int s0 = (sbeg < 1) ? 1 : sbeg;
  int send = sbeg + CHLEN; if (send > s2) send = s2;   // mask never reaches >= s2
  const float* base = qd + (size_t)b * SQ * HQ + h;
  float a1 = 0.f, a2 = 0.f;
  for (int s = s0; s < send; ++s) {
    __builtin_prefetch(base + (size_t)(s + 4) * HQ, 0, 0); // global_prefetch_b8
    float v = base[(size_t)s * HQ];
    if (s < s1) a1 += v;
    else if (s > s1) a2 += v;
  }
  partQ[((size_t)b * NCHUNK + c) * HQ + h] = a1;
  partD[((size_t)b * NCHUNK + c) * HQ + h] = a2;
}

// Pass 2: fixed-order reduction (deterministic across replays).
__global__ void pool_reduce(const float* __restrict__ partQ, const float* __restrict__ partD,
                            const int* __restrict__ sep1, const int* __restrict__ sep2,
                            float* __restrict__ Vq, float* __restrict__ Vd) {
  int b = blockIdx.x;
  int h = threadIdx.x;
  float q = 0.f, d = 0.f;
  #pragma unroll
  for (int c = 0; c < NCHUNK; ++c) {
    q += partQ[((size_t)b * NCHUNK + c) * HQ + h];
    d += partD[((size_t)b * NCHUNK + c) * HQ + h];
  }
  Vq[b * HQ + h] = q / (float)(sep1[b] - 1);
  Vd[b * HQ + h] = d / (float)(sep2[b] - sep1[b] - 1);
}

// ----------------- 2. pack node features X = [V_s;V_t] -> f16 --------------
__global__ void pack_x_f16(const float* __restrict__ Vs, const float* __restrict__ Vt,
                           _Float16* __restrict__ X) {
  int idx = blockIdx.x * blockDim.x + threadIdx.x;   // NNODES*HQ elements
  int n = idx / HQ, h = idx - n * HQ;
  int b = n >> 3, i = n & 7;
  float v = (i < 4) ? Vs[((size_t)b * 4 + i) * HQ + h]
                    : Vt[((size_t)b * 4 + (i - 4)) * HQ + h];
  X[idx] = (_Float16)v;
}

// -------------------- 3. transpose fp32 [K,N] -> f16 [N,K] -----------------
__global__ void transpose_to_f16(const float* __restrict__ W, _Float16* __restrict__ Wt,
                                 int K, int N) {
  int idx = blockIdx.x * blockDim.x + threadIdx.x;
  if (idx >= K * N) return;
  int k = idx / N, n = idx - k * N;
  Wt[(size_t)n * K + k] = (_Float16)W[idx];
}

// ------------------------- 4. WMMA tiled GEMM ------------------------------
// C[M,N] = A[M,K] (f16 row-major) x Bt[N,K]^T (f16, N-major rows of K).
// One wave computes a 16x32 strip: one A fragment feeds two
// v_wmma_f32_16x16x32_f16 per k-step. act: 0=none, 1=tanh. bias nullable.
__global__ void gemm_wmma(const _Float16* __restrict__ A, const _Float16* __restrict__ Bt,
                          float* __restrict__ C, const float* __restrict__ bias,
                          int M, int N, int K, int act) {
  int wid  = (blockIdx.x * blockDim.x + threadIdx.x) >> 5;  // wave-uniform
  int lane = threadIdx.x & 31;
  int tilesN2 = N >> 5;                // 32-wide N strips
  int tm = wid / tilesN2;
  int m0 = tm << 4;
  int n0 = (wid - tm * tilesN2) << 5;
  if (m0 >= M) return;                 // whole-wave exit; EXEC stays all-1s
  int lo = lane & 15, hi = lane >> 4;

  // A fragment: lanes 0-15 carry K {k..k+7, k+16..k+23}, lanes 16-31 shift by 8
  const _Float16* arow  = A  + (size_t)(m0 + lo) * K + hi * 8;
  // B fragment: N = n0+lo, lanes 0-15 carry K {k..k+7, k+16..k+23}, lanes 16-31 shift by 8
  const _Float16* brow0 = Bt + (size_t)(n0 + lo) * K + hi * 8;
  const _Float16* brow1 = Bt + (size_t)(n0 + 16 + lo) * K + hi * 8;

  v8f acc0 = {}, acc1 = {};
  for (int k = 0; k < K; k += 32) {
    v8h a0 = *(const v8h*)(arow + k);
    v8h a1 = *(const v8h*)(arow + k + 16);
    v16h a = __builtin_shufflevector(a0, a1, 0,1,2,3,4,5,6,7,8,9,10,11,12,13,14,15);
    v16h b0 = __builtin_shufflevector(*(const v8h*)(brow0 + k), *(const v8h*)(brow0 + k + 16), 0,1,2,3,4,5,6,7,8,9,10,11,12,13,14,15);
    v16h b1 = __builtin_shufflevector(*(const v8h*)(brow1 + k), *(const v8h*)(brow1 + k + 16), 0,1,2,3,4,5,6,7,8,9,10,11,12,13,14,15);
    acc0 = __builtin_amdgcn_wmma_f32_16x16x32_f16(false, a, false, b0,
                                                  (short)0, acc0, false, false);
    acc1 = __builtin_amdgcn_wmma_f32_16x16x32_f16(false, a, false, b1,
                                                  (short)0, acc1, false, false);
  }
  float bn0 = bias ? bias[n0 + lo] : 0.0f;
  float bn1 = bias ? bias[n0 + 16 + lo] : 0.0f;
  // C/D layout: VGPR r -> M = m0 + r + hi*8, N = n0 + lo (and +16 for acc1)
  float* cbase = C + (size_t)(m0 + hi * 8) * N + n0 + lo;
  #pragma unroll
  for (int r = 0; r < 8; ++r) {
    float v0 = acc0[r] + bn0;
    float v1 = acc1[r] + bn1;
    if (act == 1) { v0 = tanhf(v0); v1 = tanhf(v1); }
    cbase[(size_t)r * N]      = v0;
    cbase[(size_t)r * N + 16] = v1;
  }
}

// -------------------- 5. attention scalars a_src/a_dst ---------------------
__global__ void att_scores(const float* __restrict__ Hmat,
                           const float* __restrict__ att_s, const float* __restrict__ att_d,
                           float* __restrict__ asrc, float* __restrict__ adst,
                           int nodes, int heads, int C) {
  int gw = (blockIdx.x * blockDim.x + threadIdx.x) >> 5;
  int lane = threadIdx.x & 31;
  if (gw >= nodes * heads) return;
  int node = gw / heads, head = gw - node * heads;
  const float* hrow = Hmat + ((size_t)node * heads + head) * C;
  float s1 = 0.f, s2 = 0.f;
  for (int c = lane; c < C; c += 32) {
    float hv = hrow[c];
    s1 += hv * att_s[head * C + c];
    s2 += hv * att_d[head * C + c];
  }
  s1 = wave_reduce_sum(s1);
  s2 = wave_reduce_sum(s2);
  if (lane == 0) { asrc[gw] = s1; adst[gw] = s2; }
}

// ---------------- 6. GAT layer 1: softmax-agg + bias + relu -> f16 ---------
__global__ void gat1_agg(const float* __restrict__ Hmat, const float* __restrict__ asrc,
                         const float* __restrict__ adst, const float* __restrict__ bias,
                         _Float16* __restrict__ X2) {
  const int kInDeg[8]    = {5,3,3,3,5,3,3,3};
  const int kInSrc[8][5] = {{4,1,2,3,0},{0,5,1,0,0},{0,6,2,0,0},{0,7,3,0,0},
                            {0,5,6,7,4},{1,4,5,0,0},{2,4,6,0,0},{3,4,7,0,0}};
  int g = blockIdx.x;                 // 64 graphs
  int t = threadIdx.x;                // 512 = 4 heads x 128 ch
  int head = t >> 7, c = t & 127;
  #pragma unroll
  for (int d = 0; d < 8; ++d) {
    const int deg = kInDeg[d];
    int dn = g * NPG_ + d;
    float ad = adst[dn * HEADS1_ + head];
    float w[5];
    float mx = -1e30f;
    #pragma unroll
    for (int e = 0; e < deg; ++e) {
      int sn = g * NPG_ + kInSrc[d][e];
      float al = asrc[sn * HEADS1_ + head] + ad;
      al = (al > 0.f) ? al : 0.2f * al;            // leaky_relu(0.2)
      w[e] = al;
      mx = fmaxf(mx, al);
    }
    float den = 0.f;
    #pragma unroll
    for (int e = 0; e < deg; ++e) { w[e] = __expf(w[e] - mx); den += w[e]; }
    float inv = 1.0f / (den + 1e-16f);
    float acc = 0.f;
    #pragma unroll
    for (int e = 0; e < deg; ++e) {
      int sn = g * NPG_ + kInSrc[d][e];
      acc += w[e] * Hmat[(size_t)sn * H1DIM + head * C1_ + c];
    }
    acc = fmaxf(acc * inv + bias[head * C1_ + c], 0.f);   // +bias, relu
    X2[(size_t)dn * H1DIM + head * C1_ + c] = (_Float16)acc;
  }
}

// -------- 7. GAT layer 2 (1 head) agg + bias, fused global mean pool -------
__global__ void gat2_agg_mean(const float* __restrict__ H2, const float* __restrict__ asrc,
                              const float* __restrict__ adst, const float* __restrict__ b2,
                              float* __restrict__ Vkg) {
  const int kInDeg[8]    = {5,3,3,3,5,3,3,3};
  const int kInSrc[8][5] = {{4,1,2,3,0},{0,5,1,0,0},{0,6,2,0,0},{0,7,3,0,0},
                            {0,5,6,7,4},{1,4,5,0,0},{2,4,6,0,0},{3,4,7,0,0}};
  int g = blockIdx.x;          // 64 graphs
  int c = threadIdx.x;         // 768 channels
  float accg = 0.f;
  #pragma unroll
  for (int d = 0; d < 8; ++d) {
    const int deg = kInDeg[d];
    int dn = g * NPG_ + d;
    float ad = adst[dn];
    float w[5];
    float mx = -1e30f;
    #pragma unroll
    for (int e = 0; e < deg; ++e) {
      int sn = g * NPG_ + kInSrc[d][e];
      float al = asrc[sn] + ad;
      al = (al > 0.f) ? al : 0.2f * al;
      w[e] = al;
      mx = fmaxf(mx, al);
    }
    float den = 0.f;
    #pragma unroll
    for (int e = 0; e < deg; ++e) { w[e] = __expf(w[e] - mx); den += w[e]; }
    float inv = 1.0f / (den + 1e-16f);
    float acc = 0.f;
    #pragma unroll
    for (int e = 0; e < deg; ++e) {
      int sn = g * NPG_ + kInSrc[d][e];
      acc += w[e] * H2[(size_t)sn * HQ + c];
    }
    accg += acc * inv;
  }
  Vkg[(size_t)g * HQ + c] = accg * 0.125f + b2[c];   // mean over 8 nodes (+bias)
}

// -------------------- 8. pack [V_q | V_kg] -> f16 --------------------------
__global__ void fusion_pack(const float* __restrict__ Vq, const float* __restrict__ Vkg,
                            _Float16* __restrict__ Af16) {
  int idx = blockIdx.x * blockDim.x + threadIdx.x;   // 64*1536
  int r = idx / 1536, c = idx - r * 1536;
  float v = (c < HQ) ? Vq[(size_t)r * HQ + c] : Vkg[(size_t)r * HQ + (c - HQ)];
  Af16[idx] = (_Float16)v;
}

// --------------------- 9. cosine similarity pieces -------------------------
__global__ void rownorm_inv(const float* __restrict__ X, float* __restrict__ inv,
                            int rows, int cols) {
  int gw = (blockIdx.x * blockDim.x + threadIdx.x) >> 5;
  int lane = threadIdx.x & 31;
  if (gw >= rows) return;
  const float* r = X + (size_t)gw * cols;
  float s = 0.f;
  for (int c = lane; c < cols; c += 32) s += r[c] * r[c];
  s = wave_reduce_sum(s);
  if (lane == 0) inv[gw] = 1.0f / fmaxf(sqrtf(s), 1e-8f);
}

__global__ void cos_scores(const float* __restrict__ Q, const float* __restrict__ D,
                           const float* __restrict__ invq, const float* __restrict__ invd,
                           float* __restrict__ out) {
  int gw = (blockIdx.x * blockDim.x + threadIdx.x) >> 5;   // 64*64 waves
  int lane = threadIdx.x & 31;
  if (gw >= BQ * BQ) return;
  int i = gw >> 6, j = gw & 63;
  const float* qr = Q + (size_t)i * HQ;
  const float* dr = D + (size_t)j * HQ;
  float s = 0.f;
  for (int c = lane; c < HQ; c += 32) s += qr[c] * dr[c];
  s = wave_reduce_sum(s);
  if (lane == 0) out[gw] = s * invq[i] * invd[j];
}

// ---------------------------------------------------------------------------

extern "C" void kernel_launch(void* const* d_in, const int* in_sizes, int n_in,
                              void* d_out, int out_size, void* d_ws, size_t ws_size,
                              hipStream_t stream) {
  const float* qd  = (const float*)d_in[0];
  const float* Vs  = (const float*)d_in[1];
  const float* Vt  = (const float*)d_in[2];
  const float* W1  = (const float*)d_in[3];
  const float* as1 = (const float*)d_in[4];
  const float* ad1 = (const float*)d_in[5];
  const float* b1  = (const float*)d_in[6];
  const float* W2  = (const float*)d_in[7];
  const float* as2 = (const float*)d_in[8];
  const float* ad2 = (const float*)d_in[9];
  const float* b2  = (const float*)d_in[10];
  const float* Wf  = (const float*)d_in[11];
  const float* bf  = (const float*)d_in[12];
  const int* sep1  = (const int*)d_in[13];
  const int* sep2  = (const int*)d_in[14];
  float* scores = (float*)d_out;

  char* ws = (char*)d_ws;
  size_t off = 0;
  auto alloc = [&](size_t bytes) -> void* {
    void* p = ws + off;
    off = (off + bytes + 255) & ~(size_t)255;
    return p;
  };

  float*    partQ = (float*)   alloc((size_t)BQ * NCHUNK * HQ * 4);
  float*    partD = (float*)   alloc((size_t)BQ * NCHUNK * HQ * 4);
  float*    Vq    = (float*)   alloc((size_t)BQ * HQ * 4);
  float*    Vd    = (float*)   alloc((size_t)BQ * HQ * 4);
  _Float16* Xf16  = (_Float16*)alloc((size_t)NNODES * HQ * 2);
  _Float16* W1t   = (_Float16*)alloc((size_t)H1DIM * HQ * 2);       // [512,768]
  float*    Hm1   = (float*)   alloc((size_t)NNODES * H1DIM * 4);   // [512,512]
  float*    asrc1 = (float*)   alloc((size_t)NNODES * HEADS1_ * 4);
  float*    adst1 = (float*)   alloc((size_t)NNODES * HEADS1_ * 4);
  _Float16* X2f16 = (_Float16*)alloc((size_t)NNODES * H1DIM * 2);
  _Float16* W2t   = (_Float16*)alloc((size_t)HQ * H1DIM * 2);       // [768,512]
  float*    Hm2   = (float*)   alloc((size_t)NNODES * HQ * 4);      // [512,768]
  float*    asrc2 = (float*)   alloc((size_t)NNODES * 4);
  float*    adst2 = (float*)   alloc((size_t)NNODES * 4);
  float*    Vkg   = (float*)   alloc((size_t)BQ * HQ * 4);
  _Float16* Af16  = (_Float16*)alloc((size_t)BQ * 2 * HQ * 2);      // [64,1536]
  _Float16* Wft   = (_Float16*)alloc((size_t)HQ * 2 * HQ * 2);      // [768,1536]
  float*    Vqkg  = (float*)   alloc((size_t)BQ * HQ * 4);
  float*    invq  = (float*)   alloc(BQ * 4);
  float*    invd  = (float*)   alloc(BQ * 4);
  (void)off; (void)ws_size; (void)in_sizes; (void)n_in; (void)out_size;

  // 1) pooling (bandwidth-dominant stage): 512 blocks, then fixed-order reduce
  pool_partial<<<dim3(BQ, NCHUNK), HQ, 0, stream>>>(qd, sep1, sep2, partQ, partD);
  pool_reduce<<<BQ, HQ, 0, stream>>>(partQ, partD, sep1, sep2, Vq, Vd);

  // 2) pack node features + transpose/convert the three weight matrices
  pack_x_f16<<<(NNODES * HQ) / 256, 256, 0, stream>>>(Vs, Vt, Xf16);
  transpose_to_f16<<<(HQ * H1DIM) / 256, 256, 0, stream>>>(W1, W1t, HQ, H1DIM);
  transpose_to_f16<<<(H1DIM * HQ) / 256, 256, 0, stream>>>(W2, W2t, H1DIM, HQ);
  transpose_to_f16<<<(2 * HQ * HQ) / 256, 256, 0, stream>>>(Wf, Wft, 2 * HQ, HQ);

  // 3) GAT1: H1 = X @ W1 (512x768 x 768x512): 32x16 strips = 512 waves
  gemm_wmma<<<64, 256, 0, stream>>>(Xf16, W1t, Hm1, nullptr, NNODES, H1DIM, HQ, 0);
  att_scores<<<256, 256, 0, stream>>>(Hm1, as1, ad1, asrc1, adst1, NNODES, HEADS1_, C1_);
  gat1_agg<<<BQ, H1DIM, 0, stream>>>(Hm1, asrc1, adst1, b1, X2f16);

  // 4) GAT2: H2 = X2 @ W2 (512x512 x 512x768): 32x24 strips = 768 waves
  gemm_wmma<<<96, 256, 0, stream>>>(X2f16, W2t, Hm2, nullptr, NNODES, HQ, H1DIM, 0);
  att_scores<<<64, 256, 0, stream>>>(Hm2, as2, ad2, asrc2, adst2, NNODES, 1, HQ);
  gat2_agg_mean<<<BQ, HQ, 0, stream>>>(Hm2, asrc2, adst2, b2, Vkg);

  // 5) fusion: V_q_kg = tanh([Vq|Vkg] @ Wf + bf): 4x24 strips = 96 waves
  fusion_pack<<<(BQ * 2 * HQ) / 256, 256, 0, stream>>>(Vq, Vkg, Af16);
  gemm_wmma<<<12, 256, 0, stream>>>(Af16, Wft, Vqkg, bf, BQ, HQ, 2 * HQ, 1);

  // 6) cosine similarity matrix
  rownorm_inv<<<8, 256, 0, stream>>>(Vqkg, invq, BQ, HQ);
  rownorm_inv<<<8, 256, 0, stream>>>(Vd, invd, BQ, HQ);
  cos_scores<<<512, 256, 0, stream>>>(Vqkg, Vd, invq, invd, scores);
}
